// Encoder_RegLSTM_70394513981508
// MI455X (gfx1250) — matrix operand order repaired and checked by reading, not compiled, not run on hardware
//
#include <hip/hip_runtime.h>
#include <cstdint>
#include <cstddef>

#define USE_ASYNC 1

#define B_  64
#define S_  2048
#define E_  256
#define H_  256
#define G4  1024   // 4*H
#define KC  512    // E + H (concatenated K dim)

typedef __bf16   bf16_t;
typedef bf16_t   v16bf  __attribute__((ext_vector_type(16)));
typedef float    v8f    __attribute__((ext_vector_type(8)));
typedef unsigned u32x4  __attribute__((ext_vector_type(4)));

struct F2 { u32x4 a, b; };   // 32 bytes -> one v16bf fragment

__device__ __forceinline__ unsigned f2bf(float f) {
    unsigned u = __float_as_uint(f);
    u += 0x7FFFu + ((u >> 16) & 1u);      // round-to-nearest-even
    return u >> 16;
}
__device__ __forceinline__ float sigmoidf_(float x) { return 1.0f / (1.0f + __expf(-x)); }

// ---- CDNA5 async global->LDS copy (64 bytes per lane = 4x b128), ASYNCcnt tracked ----
__device__ __forceinline__ void copy64_async(float* dst_lds, const float* src) {
#if USE_ASYNC
    unsigned lds_off = (unsigned)(uintptr_t)dst_lds;      // low 32 bits of flat addr = LDS offset
    unsigned long long g = (unsigned long long)(uintptr_t)src;
#pragma unroll
    for (int i = 0; i < 4; ++i) {
        asm volatile("global_load_async_to_lds_b128 %0, %1, off"
                     :: "v"(lds_off + 16u * i), "v"(g + 16ull * i)
                     : "memory");
    }
#else
#pragma unroll
    for (int i = 0; i < 16; ++i) dst_lds[i] = src[i];
#endif
}
__device__ __forceinline__ void wait_async() {
#if USE_ASYNC
    asm volatile("s_wait_asynccnt 0x0" ::: "memory");
#endif
}

// A fragment (16x32 bf16) from LDS xh: lane<16 -> row=lane, K {k0..k0+7, k0+16..k0+23}
//                                      lane>=16 -> row=lane-16, K {k0+8..15, k0+24..31}
__device__ __forceinline__ v16bf load_a_frag(const unsigned short* xh, int lane, int k0) {
    int row  = lane & 15;
    int off1 = k0 + ((lane >> 4) << 3);
    F2 t { *reinterpret_cast<const u32x4*>(xh + row * KC + off1),
           *reinterpret_cast<const u32x4*>(xh + row * KC + off1 + 16) };
    return __builtin_bit_cast(v16bf, t);
}

// B fragment (32x16 bf16) from row-major Wc[g][k]: lane<16 -> col g0+lane, K {k0..k0+7, k0+16..k0+23};
//                                                  lane>=16 -> col g0+lane-16, K {k0+8..15, k0+24..31}
__device__ __forceinline__ v16bf load_b_frag(const unsigned short* __restrict__ Wc,
                                             int g0, int lane, int k0) {
    int col = g0 + (lane & 15);
    int off = k0 + ((lane >> 4) << 3);
    const u32x4* p = reinterpret_cast<const u32x4*>(Wc + (size_t)col * KC + off);
    F2 t { p[0], *reinterpret_cast<const u32x4*>(Wc + (size_t)col * KC + off + 16) };
    return __builtin_bit_cast(v16bf, t);
}

// ---- weight prep: Wc[g][0:256]=W_ih[g], Wc[g][256:512]=W_hh[g] (bf16); biasc = b_ih+b_hh ----
__global__ void prep_weights_kernel(const float* __restrict__ W_ih, const float* __restrict__ W_hh,
                                    const float* __restrict__ b_ih, const float* __restrict__ b_hh,
                                    unsigned short* __restrict__ Wc, float* __restrict__ biasc) {
    int idx = blockIdx.x * blockDim.x + threadIdx.x;
    if (idx < G4 * KC) {
        int g = idx / KC, k = idx - g * KC;
        float w = (k < E_) ? W_ih[g * E_ + k] : W_hh[g * H_ + (k - E_)];
        Wc[idx] = (unsigned short)f2bf(w);
    }
    if (idx < G4) biasc[idx] = b_ih[idx] + b_hh[idx];
}

// ---- persistent LSTM scan: 4 WGs x 16 batch rows; 8 waves x 32 hidden channels each ----
__global__ __launch_bounds__(256)
void lstm_scan_kernel(const int*   __restrict__ tokens,
                      const float* __restrict__ mask,
                      const float* __restrict__ embed,
                      const unsigned short* __restrict__ Wc,
                      const float* __restrict__ biasc,
                      float* __restrict__ out) {
    __shared__ __align__(16) unsigned short xh[16][KC];    // bf16 [x_t | h], 16 KB
    __shared__ __align__(16) float xraw[2][16][E_];        // fp32 gather double-buffer, 32 KB
    __shared__ float sbias[G4];                            // 4 KB

    const int tid  = threadIdx.x;
    const int lane = tid & 31;
    const int wave = tid >> 5;
    const int b0   = blockIdx.x * 16;

    for (int i = tid; i < G4; i += 256) sbias[i] = biasc[i];
    for (int i = tid; i < 16 * H_; i += 256) xh[i >> 8][E_ + (i & 255)] = 0;  // h0 = 0

    float creg[16];                                        // cell state, registers only
#pragma unroll
    for (int i = 0; i < 16; ++i) creg[i] = 0.f;

    {   // prefetch x_0
        int row = tid >> 4, seg = tid & 15;
        int tok = tokens[(b0 + row) * S_];
        copy64_async(&xraw[0][row][seg * 16], embed + (size_t)tok * E_ + seg * 16);
    }

    for (int t = 0; t < S_; ++t) {
        wait_async();
        __syncthreads();                                   // xraw[t&1] landed; prior h writes visible
        {   // convert gathered fp32 x_t -> bf16 pairs into xh[:,0:256] (packed b32 stores)
            const float2* xr2 = reinterpret_cast<const float2*>(&xraw[t & 1][0][0]);
            unsigned* xh32 = reinterpret_cast<unsigned*>(&xh[0][0]);
            for (int i = tid; i < 16 * E_ / 2; i += 256) {
                int r = i >> 7, kp = i & 127;              // kp: u32 index within the x half-row
                float2 v = xr2[i];
                xh32[r * (KC / 2) + kp] = f2bf(v.x) | (f2bf(v.y) << 16);
            }
        }
        __syncthreads();
        if (t + 1 < S_) {                                  // overlap next gather with WMMAs
            int row = tid >> 4, seg = tid & 15;
            int tok = tokens[(b0 + row) * S_ + (t + 1)];
            copy64_async(&xraw[(t + 1) & 1][row][seg * 16],
                         embed + (size_t)tok * E_ + seg * 16);
        }

        // GEMM: [16 x 512] x [512 x 1024] restricted to this wave's 8 gate-column tiles
        v8f acc[8];
#pragma unroll
        for (int nt = 0; nt < 8; ++nt) acc[nt] = v8f{0.f, 0.f, 0.f, 0.f, 0.f, 0.f, 0.f, 0.f};
        for (int kk = 0; kk < KC; kk += 32) {
            v16bf a = load_a_frag(&xh[0][0], lane, kk);
#pragma unroll
            for (int nt = 0; nt < 8; ++nt) {
                int g0 = (nt >> 1) * H_ + wave * 32 + (nt & 1) * 16;  // gate (nt>>1), sub-tile (nt&1)
                v16bf b = load_b_frag(Wc, g0, lane, kk);
                acc[nt] = __builtin_amdgcn_wmma_f32_16x16x32_bf16(
                    false, a, false, b, (short)0, acc[nt], false, false);
            }
        }
        __syncthreads();                                   // all xh reads done before h update

        // cell update in registers: i=acc[0|1], f=acc[2|3], g=acc[4|5], o=acc[6|7]
        const int hi = lane >> 4;
        float mrow[8];
#pragma unroll
        for (int q = 0; q < 8; ++q)
            mrow[q] = mask[(size_t)(b0 + q + 8 * hi) * S_ + t];
#pragma unroll
        for (int jt = 0; jt < 2; ++jt) {
            int j = wave * 32 + jt * 16 + (lane & 15);
            float bi = sbias[j], bff = sbias[H_ + j], bg = sbias[2 * H_ + j], bo = sbias[3 * H_ + j];
#pragma unroll
            for (int q = 0; q < 8; ++q) {
                int r = q + 8 * hi;
                float ip = acc[0 + jt][q] + bi;
                float fp = acc[2 + jt][q] + bff;
                float gp = acc[4 + jt][q] + bg;
                float op = acc[6 + jt][q] + bo;
                float c = creg[jt * 8 + q];
                c = sigmoidf_(fp) * c + sigmoidf_(ip) * tanhf(gp);
                float h = sigmoidf_(op) * tanhf(c);
                creg[jt * 8 + q] = c;
                xh[r][E_ + j] = (unsigned short)f2bf(h);
                size_t bs = (size_t)(b0 + r) * S_ + t;
                __builtin_nontemporal_store(h * mrow[q], &out[bs * H_ + j]);  // write-once stream
            }
        }
        // loop-top barrier orders these h writes against the next step's reads
    }
}

extern "C" void kernel_launch(void* const* d_in, const int* in_sizes, int n_in,
                              void* d_out, int out_size, void* d_ws, size_t ws_size,
                              hipStream_t stream) {
    const int*   tokens = (const int*)  d_in[0];
    const float* mask   = (const float*)d_in[1];
    // d_in[2] = len_seq: sort-by-length then unsort is the identity for a per-row LSTM -> unused
    const float* embed  = (const float*)d_in[3];
    const float* W_ih   = (const float*)d_in[4];
    const float* W_hh   = (const float*)d_in[5];
    const float* b_ih   = (const float*)d_in[6];
    const float* b_hh   = (const float*)d_in[7];
    float* out = (float*)d_out;

    unsigned short* Wc    = (unsigned short*)d_ws;                       // 1 MB bf16
    float*          biasc = (float*)((char*)d_ws + (size_t)G4 * KC * 2); // 4 KB

    int total = G4 * KC;
    prep_weights_kernel<<<(total + 255) / 256, 256, 0, stream>>>(W_ih, W_hh, b_ih, b_hh, Wc, biasc);
    lstm_scan_kernel<<<4, 256, 0, stream>>>(tokens, mask, embed, Wc, biasc, out);
}
